// HeteroGNNEncoder_Subgraph_44702019616951
// MI455X (gfx1250) — matrix operand order repaired and checked by reading, not compiled, not run on hardware
//
#include <hip/hip_runtime.h>
#include <cstdint>
#include <cstddef>

// Problem dimensions (fixed by the reference).
#define N_SNN 100000
#define N_NOC 10000
#define E_SS  500000
#define E_NN  100000
#define E_SN  100000
#define E_NS  100000
#define FDIM  128     // H*C for layer1 (4*32) and layer2 (1*128)
#define D_EDGE 16

typedef __bf16 v16bf __attribute__((ext_vector_type(16)));
typedef float  v8f   __attribute__((ext_vector_type(8)));

__device__ __forceinline__ unsigned f2ord(float f) {
  unsigned u = __float_as_uint(f);
  return (u & 0x80000000u) ? ~u : (u | 0x80000000u);
}

// ---------------------------------------------------------------------------
// GEMM: C[M,128] = A[M,K] @ W[K,128] + bias   (bf16 WMMA, f32 accumulate)
// One wave32 computes one 16x16 tile.
// PRECONDITION (guaranteed by all call sites): M % 16 == 0, K % 32 == 0.
// Fragment layouts per CDNA5 ISA §7.12.2:
//   A 16x32 bf16 : lanes 0-15 hold row r, K {hi*8+0..7, 16+hi*8+0..7}
//                  -> two contiguous 8-float runs => 4x b128 loads.
//   B 32x16 bf16 : lane half hi holds K {hi*8+0..7, 16+hi*8+0..7} of column `col`.
//   C/D 16x16 f32: VGPR v -> row v + 8*hi, col = lane&15.
// ---------------------------------------------------------------------------
__global__ void gemm_bias_wmma(const float* __restrict__ A,
                               const float* __restrict__ W,
                               const float* __restrict__ bias,
                               float* __restrict__ Cout, int M, int K) {
  const int wave  = threadIdx.x >> 5;
  const int lane  = threadIdx.x & 31;
  const int mtile = blockIdx.x * (blockDim.x >> 5) + wave;
  const int ntile = blockIdx.y;              // 0..7 (N = 128)
  if (mtile * 16 >= M) return;               // uniform per wave (EXEC stays full)
  const int r   = lane & 15;
  const int hi  = lane >> 4;                 // lane half
  const int col = ntile * 16 + r;            // B/C/D column for this lane
  const float* Arow = A + (size_t)(mtile * 16 + r) * K + hi * 8;

  v8f acc = {0.f, 0.f, 0.f, 0.f, 0.f, 0.f, 0.f, 0.f};
  for (int k0 = 0; k0 < K; k0 += 32) {
    const float4 a0 = *(const float4*)(Arow + k0);        // K = hi*8 + 0..3
    const float4 a1 = *(const float4*)(Arow + k0 + 4);    // K = hi*8 + 4..7
    const float4 a2 = *(const float4*)(Arow + k0 + 16);   // K = 16+hi*8 + 0..3
    const float4 a3 = *(const float4*)(Arow + k0 + 20);   // K = 16+hi*8 + 4..7
    v16bf a, b;
    a[0]  = (__bf16)a0.x; a[1]  = (__bf16)a0.y;
    a[2]  = (__bf16)a0.z; a[3]  = (__bf16)a0.w;
    a[4]  = (__bf16)a1.x; a[5]  = (__bf16)a1.y;
    a[6]  = (__bf16)a1.z; a[7]  = (__bf16)a1.w;
    a[8]  = (__bf16)a2.x; a[9]  = (__bf16)a2.y;
    a[10] = (__bf16)a2.z; a[11] = (__bf16)a2.w;
    a[12] = (__bf16)a3.x; a[13] = (__bf16)a3.y;
    a[14] = (__bf16)a3.z; a[15] = (__bf16)a3.w;
#pragma unroll
    for (int v = 0; v < 8; ++v) {
      const int kb = k0 + hi * 8 + 2 * v + (v >= 4 ? 8 : 0);
      b[2 * v]     = (__bf16)W[(size_t)kb * FDIM + col];
      b[2 * v + 1] = (__bf16)W[(size_t)(kb + 1) * FDIM + col];
    }
    acc = __builtin_amdgcn_wmma_f32_16x16x32_bf16(false, a, false, b,
                                                  (short)0, acc, false, false);
  }
  const float bv = bias ? bias[col] : 0.f;
#pragma unroll
  for (int v = 0; v < 8; ++v)
    Cout[(size_t)(mtile * 16 + v + hi * 8) * FDIM + col] = acc[v] + bv;
}

// ---------------------------------------------------------------------------
// Edge phase A: e[E,H] = leaky(xl[src]+xr[dst](+attr@We)) . att   + seg-max
// One wave per edge, 4 features per lane, shfl_xor reduce per head group.
// ---------------------------------------------------------------------------
template <int H, bool EATTR>
__global__ void edge_attn_logits(const int* __restrict__ ei, int E,
                                 const float* __restrict__ xl,
                                 const float* __restrict__ xr,
                                 const float* __restrict__ eattr,
                                 const float* __restrict__ We,
                                 const float* __restrict__ att,
                                 float* __restrict__ ebuf,
                                 unsigned* __restrict__ emax) {
  const int e = blockIdx.x * (blockDim.x >> 5) + (threadIdx.x >> 5);
  if (e >= E) return;
  const int lane = threadIdx.x & 31;
  const int src = ei[e];
  const int dst = ei[E + e];
  const int f = lane * 4;
  const float4 xlv = *(const float4*)(xl + (size_t)src * FDIM + f);
  const float4 xrv = *(const float4*)(xr + (size_t)dst * FDIM + f);
  float z0 = xlv.x + xrv.x, z1 = xlv.y + xrv.y;
  float z2 = xlv.z + xrv.z, z3 = xlv.w + xrv.w;
  if constexpr (EATTR) {                       // fold edge_attr @ We on the fly
    const float* ap = eattr + (size_t)e * D_EDGE;
#pragma unroll
    for (int k = 0; k < D_EDGE; ++k) {
      const float a = ap[k];
      const float4 wv = *(const float4*)(We + (size_t)k * FDIM + f);
      z0 += a * wv.x; z1 += a * wv.y; z2 += a * wv.z; z3 += a * wv.w;
    }
  }
  z0 = (z0 >= 0.f) ? z0 : 0.2f * z0;           // leaky_relu slope 0.2
  z1 = (z1 >= 0.f) ? z1 : 0.2f * z1;
  z2 = (z2 >= 0.f) ? z2 : 0.2f * z2;
  z3 = (z3 >= 0.f) ? z3 : 0.2f * z3;
  const float4 av = *(const float4*)(att + f);
  float s = z0 * av.x + z1 * av.y + z2 * av.z + z3 * av.w;

  constexpr int GROUP = 32 / H;                // lanes per head (8 or 32)
#pragma unroll
  for (int m = 1; m < GROUP; m <<= 1) s += __shfl_xor(s, m, 32);
  if ((lane & (GROUP - 1)) == 0) {
    const int h = lane / GROUP;
    ebuf[(size_t)e * H + h] = s;
    atomicMax(&emax[(size_t)dst * H + h], f2ord(s));  // ordered-uint float max
  }
}

// ---------------------------------------------------------------------------
// Edge phase B: ee = exp(e - emax[dst]); den[dst] += ee
// ---------------------------------------------------------------------------
__global__ void edge_softmax_num(const int* __restrict__ ei, int E, int H,
                                 const float* __restrict__ emax,
                                 float* __restrict__ ebuf,
                                 float* __restrict__ den) {
  const int t = blockIdx.x * blockDim.x + threadIdx.x;
  if (t >= E * H) return;
  const int e = t / H;
  const int h = t - e * H;
  const int dst = ei[E + e];
  const float ee = expf(ebuf[t] - emax[dst * H + h]);
  ebuf[t] = ee;
  atomicAdd(&den[dst * H + h], ee);
}

// ---------------------------------------------------------------------------
// Edge phase C: out[dst] += (ee/(den+1e-16)) * xl[src]   (atomic scatter)
// ---------------------------------------------------------------------------
template <int H>
__global__ void edge_aggregate(const int* __restrict__ ei, int E,
                               const float* __restrict__ xl,
                               const float* __restrict__ ebuf,
                               const float* __restrict__ den,
                               float* __restrict__ out) {
  const int e = blockIdx.x * (blockDim.x >> 5) + (threadIdx.x >> 5);
  if (e >= E) return;
  const int lane = threadIdx.x & 31;
  const int src = ei[e];
  const int dst = ei[E + e];
  const int f = lane * 4;
  constexpr int GROUP = 32 / H;
  const int h = lane / GROUP;
  const float ee    = ebuf[(size_t)e * H + h];
  const float alpha = ee / (den[(size_t)dst * H + h] + 1e-16f);
  const float4 xlv = *(const float4*)(xl + (size_t)src * FDIM + f);
  float* o = out + (size_t)dst * FDIM + f;
  atomicAdd(o + 0, alpha * xlv.x);
  atomicAdd(o + 1, alpha * xlv.y);
  atomicAdd(o + 2, alpha * xlv.z);
  atomicAdd(o + 3, alpha * xlv.w);
}

// ---------------------------------------------------------------------------
// Utility kernels
// ---------------------------------------------------------------------------
__global__ void fill_u32_kernel(unsigned* p, unsigned v, int n) {
  const int t = blockIdx.x * blockDim.x + threadIdx.x;
  if (t < n) p[t] = v;
}

// ordered-uint -> float, replace non-finite (empty segment) with 0; zero den.
__global__ void finalize_max_init_den(unsigned* maxbuf, float* den, int n) {
  const int t = blockIdx.x * blockDim.x + threadIdx.x;
  if (t >= n) return;
  const unsigned u = maxbuf[t];
  float m = (u & 0x80000000u) ? __uint_as_float(u & 0x7fffffffu)
                              : __uint_as_float(~u);
  if (!__builtin_isfinite(m)) m = 0.f;
  ((float*)maxbuf)[t] = m;
  den[t] = 0.f;
}

// out[t] = b1[t%128] + b2[t%128]  (both conv biases of the hetero sum)
__global__ void init_row_bias2(float* out, const float* __restrict__ b1,
                               const float* __restrict__ b2, int total) {
  const int t = blockIdx.x * blockDim.x + threadIdx.x;
  if (t < total) out[t] = b1[t & (FDIM - 1)] + b2[t & (FDIM - 1)];
}

__global__ void relu_kernel(float* p, int n) {
  const int t = blockIdx.x * blockDim.x + threadIdx.x;
  if (t < n) p[t] = fmaxf(p[t], 0.f);
}

// ---------------------------------------------------------------------------
// Host orchestration
// ---------------------------------------------------------------------------
struct ConvParams { const float *Wl, *bl, *Wr, *br, *att, *bias, *We; };
struct Scratch { float *xl, *xr, *ebuf, *den; unsigned *emax; };

static inline int ceil_div(int a, int b) { return (a + b - 1) / b; }

static void run_gatv2(hipStream_t s,
                      const float* x_src, int Nsrc, int Ksrc,
                      const float* x_dst, int Ndst, int Kdst,
                      const int* ei, int E, const ConvParams& p, int H,
                      const float* eattr, float* out, const Scratch& w) {
  const dim3 blk(256);
  { dim3 g(ceil_div(ceil_div(Nsrc, 16), 8), 8);
    gemm_bias_wmma<<<g, blk, 0, s>>>(x_src, p.Wl, p.bl, w.xl, Nsrc, Ksrc); }
  { dim3 g(ceil_div(ceil_div(Ndst, 16), 8), 8);
    gemm_bias_wmma<<<g, blk, 0, s>>>(x_dst, p.Wr, p.br, w.xr, Ndst, Kdst); }

  const int nm = Ndst * H;
  fill_u32_kernel<<<ceil_div(nm, 256), 256, 0, s>>>(w.emax, 0x007FFFFFu, nm); // f2ord(-inf)

  const int eb = ceil_div(E, 8);     // 8 waves (edges) per 256-thread block
  if (H == 4) {
    if (eattr) edge_attn_logits<4, true ><<<eb, 256, 0, s>>>(ei, E, w.xl, w.xr, eattr, p.We, p.att, w.ebuf, w.emax);
    else       edge_attn_logits<4, false><<<eb, 256, 0, s>>>(ei, E, w.xl, w.xr, nullptr, nullptr, p.att, w.ebuf, w.emax);
  } else {
    if (eattr) edge_attn_logits<1, true ><<<eb, 256, 0, s>>>(ei, E, w.xl, w.xr, eattr, p.We, p.att, w.ebuf, w.emax);
    else       edge_attn_logits<1, false><<<eb, 256, 0, s>>>(ei, E, w.xl, w.xr, nullptr, nullptr, p.att, w.ebuf, w.emax);
  }
  finalize_max_init_den<<<ceil_div(nm, 256), 256, 0, s>>>(w.emax, w.den, nm);
  edge_softmax_num<<<ceil_div(E * H, 256), 256, 0, s>>>(ei, E, H, (const float*)w.emax, w.ebuf, w.den);
  if (H == 4) edge_aggregate<4><<<eb, 256, 0, s>>>(ei, E, w.xl, w.ebuf, w.den, out);
  else        edge_aggregate<1><<<eb, 256, 0, s>>>(ei, E, w.xl, w.ebuf, w.den, out);
}

extern "C" void kernel_launch(void* const* d_in, const int* in_sizes, int n_in,
                              void* d_out, int out_size, void* d_ws, size_t ws_size,
                              hipStream_t stream) {
  (void)in_sizes; (void)n_in; (void)out_size; (void)ws_size;

  const float* x_snn   = (const float*)d_in[0];   // [100000, 64]
  const float* x_noc   = (const float*)d_in[1];   // [10000, 32]
  const float* attr_ss = (const float*)d_in[2];   // [500000, 16]
  const int*   ei_ss   = (const int*)d_in[3];     // [2, 500000]
  const int*   ei_nn   = (const int*)d_in[4];
  const int*   ei_sn   = (const int*)d_in[5];
  const int*   ei_ns   = (const int*)d_in[6];

  int idx = 7;
  auto grab = [&](bool has_we) {
    ConvParams p;
    p.Wl   = (const float*)d_in[idx++];
    p.bl   = (const float*)d_in[idx++];
    p.Wr   = (const float*)d_in[idx++];
    p.br   = (const float*)d_in[idx++];
    p.att  = (const float*)d_in[idx++];
    p.bias = (const float*)d_in[idx++];
    p.We   = has_we ? (const float*)d_in[idx++] : nullptr;
    return p;
  };
  ConvParams ss1 = grab(true),  nn1 = grab(false), sn1 = grab(false), ns1 = grab(false);
  ConvParams ss2 = grab(true),  nn2 = grab(false), sn2 = grab(false), ns2 = grab(false);

  // Workspace carve (floats): ~170 MB total, fully re-initialized every call.
  float* ws = (float*)d_ws;
  size_t off = 0;
  float* xl   = ws + off; off += (size_t)N_SNN * FDIM;
  float* xr   = ws + off; off += (size_t)N_SNN * FDIM;
  float* hs   = ws + off; off += (size_t)N_SNN * FDIM;
  float* hn   = ws + off; off += (size_t)N_NOC * FDIM;
  float* ebuf = ws + off; off += (size_t)E_SS * 4;
  unsigned* emax = (unsigned*)(ws + off); off += (size_t)N_SNN * 4;
  float* den  = ws + off; off += (size_t)N_SNN * 4;
  Scratch w{xl, xr, ebuf, den, emax};

  // -------- Layer 1 (H=4, C=32, concat) --------
  init_row_bias2<<<ceil_div(N_SNN * FDIM, 256), 256, 0, stream>>>(hs, ss1.bias, ns1.bias, N_SNN * FDIM);
  init_row_bias2<<<ceil_div(N_NOC * FDIM, 256), 256, 0, stream>>>(hn, nn1.bias, sn1.bias, N_NOC * FDIM);
  run_gatv2(stream, x_snn, N_SNN, 64, x_snn, N_SNN, 64, ei_ss, E_SS, ss1, 4, attr_ss, hs, w);
  run_gatv2(stream, x_noc, N_NOC, 32, x_snn, N_SNN, 64, ei_ns, E_NS, ns1, 4, nullptr, hs, w);
  run_gatv2(stream, x_noc, N_NOC, 32, x_noc, N_NOC, 32, ei_nn, E_NN, nn1, 4, nullptr, hn, w);
  run_gatv2(stream, x_snn, N_SNN, 64, x_noc, N_NOC, 32, ei_sn, E_SN, sn1, 4, nullptr, hn, w);
  relu_kernel<<<ceil_div(N_SNN * FDIM, 256), 256, 0, stream>>>(hs, N_SNN * FDIM);
  relu_kernel<<<ceil_div(N_NOC * FDIM, 256), 256, 0, stream>>>(hn, N_NOC * FDIM);

  // -------- Layer 2 (H=1, C=128) -> d_out = [o_s | o_n] --------
  float* os = (float*)d_out;
  float* on = os + (size_t)N_SNN * FDIM;
  init_row_bias2<<<ceil_div(N_SNN * FDIM, 256), 256, 0, stream>>>(os, ss2.bias, ns2.bias, N_SNN * FDIM);
  init_row_bias2<<<ceil_div(N_NOC * FDIM, 256), 256, 0, stream>>>(on, nn2.bias, sn2.bias, N_NOC * FDIM);
  run_gatv2(stream, hs, N_SNN, 128, hs, N_SNN, 128, ei_ss, E_SS, ss2, 1, attr_ss, os, w);
  run_gatv2(stream, hn, N_NOC, 128, hs, N_SNN, 128, ei_ns, E_NS, ns2, 1, nullptr, os, w);
  run_gatv2(stream, hn, N_NOC, 128, hn, N_NOC, 128, ei_nn, E_NN, nn2, 1, nullptr, on, w);
  run_gatv2(stream, hs, N_SNN, 128, hn, N_NOC, 128, ei_sn, E_SN, sn2, 1, nullptr, on, w);
}
